// StackDevConv_88648124991107
// MI455X (gfx1250) — matrix operand order repaired and checked by reading, not compiled, not run on hardware
//
#include <hip/hip_runtime.h>

#define DIM    64
#define K1     128           // 2*DIM concat input to MLP1
#define WAVES  4
#define TPB    (WAVES * 32)
#define TILES_PER_WAVE 8     // 16 edges per tile -> 128 edges per wave per launch

typedef __attribute__((ext_vector_type(16))) _Float16 v16h;
typedef __attribute__((ext_vector_type(8)))  _Float16 v8h;
typedef __attribute__((ext_vector_type(2)))  _Float16 v2h;
typedef __attribute__((ext_vector_type(8)))  float    v8f;

// Order-preserving float<->int encoding so signed atomicMax == float max.
#define ENC_NEG_INF 0x807fffff   // enc(-inf)

__device__ __forceinline__ int enc_f32(float f) {
    int i = __float_as_int(f);
    return i >= 0 ? i : (i ^ 0x7fffffff);
}
__device__ __forceinline__ float dec_f32(int e) {
    int i = e >= 0 ? e : (e ^ 0x7fffffff);
    return __int_as_float(i);
}

__device__ __forceinline__ v16h cat8(v8h lo, v8h hi) {
    return __builtin_shufflevector(lo, hi, 0,1,2,3,4,5,6,7,8,9,10,11,12,13,14,15);
}
// A fragment (16x32 f16): lane holds K = [koff..koff+7] and [koff+16..koff+23]
// of its row; p points at row + koff (koff = ks*32 + kq*8).
__device__ __forceinline__ v16h load_a_frag(const _Float16* p) {
    v8h lo = *(const v8h*)(p);
    v8h hi = *(const v8h*)(p + 16);
    return cat8(lo, hi);
}
// B fragment (32x16 f16): lane holds K = [koff..koff+7] and [koff+16..koff+23] of its column; weights are
// stored [out][in] row-major so a "column" of B == a row of W.
__device__ __forceinline__ v16h load_b_frag(const _Float16* p) {
    v8h lo = *(const v8h*)(p);
    v8h hi = *(const v8h*)(p + 16);
    return cat8(lo, hi);
}
__device__ __forceinline__ v8f wmma16(v16h a, v16h b, v8f c) {
    return __builtin_amdgcn_wmma_f32_16x16x32_f16(
        /*neg_a=*/false, a, /*neg_b=*/false, b,
        /*c_mod=*/(short)0, c, /*reuse_a=*/false, /*reuse_b=*/false);
}

// ---------------------------------------------------------------------------
// One DevConv layer. Each wave owns TILES_PER_WAVE consecutive 16-edge tiles:
//   A = [x_i | x_j - x_i]          (16x128 f16, built in wave-private LDS)
//   H = relu(A @ W1^T + b1)        (16x64, via 16x v_wmma_f32_16x16x32_f16)
//   M = H @ W2^T + b2              (16x64, via  8x wmma)
//   segment-max scatter of M into aggEnc[dst] (monotone-int atomicMax).
// Weights are staged to LDS once per block and reused across all tiles.
// ---------------------------------------------------------------------------
__global__ void __launch_bounds__(TPB)
devconv_kernel(const float* __restrict__ x,
               const int*   __restrict__ srcIdx,
               const int*   __restrict__ dstIdx,
               const float* __restrict__ W1, const float* __restrict__ b1,
               const float* __restrict__ W2, const float* __restrict__ b2,
               int* __restrict__ aggEnc, int nEdges)
{
    __shared__ __align__(16) _Float16 sW1[DIM * K1];          // 16 KB, [out][in]
    __shared__ __align__(16) _Float16 sW2[DIM * DIM];         //  8 KB, [out][in]
    __shared__ float sB1[DIM];
    __shared__ float sB2[DIM];
    __shared__ __align__(16) _Float16 sA[WAVES][16 * K1];     // 16 KB
    __shared__ __align__(16) _Float16 sH[WAVES][16 * DIM];    //  8 KB
    __shared__ int sDst[WAVES][16];

    const int tid = threadIdx.x;

    // Stage weights/biases into LDS (f16 for WMMA B-operands), once per block.
    for (int i = tid; i < DIM * K1;  i += TPB) sW1[i] = (_Float16)W1[i];
    for (int i = tid; i < DIM * DIM; i += TPB) sW2[i] = (_Float16)W2[i];
    if (tid < DIM) { sB1[tid] = b1[tid]; sB2[tid] = b2[tid]; }
    __syncthreads();

    const int wave = tid >> 5;
    const int lane = tid & 31;
    const int m16  = lane & 15;   // row (A) or column (B/D) within 16
    const int kq   = lane >> 4;   // lane-group selector

    _Float16* A    = sA[wave];
    _Float16* H    = sH[wave];
    int*      dstW = sDst[wave];

    const int tilesTotal = (nEdges + 15) >> 4;
    const int tile0 = (blockIdx.x * WAVES + wave) * TILES_PER_WAVE;

    for (int ti = 0; ti < TILES_PER_WAVE; ++ti) {
        const int tile = tile0 + ti;
        if (tile >= tilesTotal) break;          // wave-uniform exit
        const int e0 = tile * 16;

        // Prefetch next tile's edge indices into cache while we compute.
        if (lane == 0) {
            __builtin_prefetch(srcIdx + e0 + 16, 0, 1);
            __builtin_prefetch(dstIdx + e0 + 16, 0, 1);
        }

        // ---- gather: build A = [x_i | x_j - x_i] as f16 rows of 128 ----
        for (int e = 0; e < 16; ++e) {
            const int eg = e0 + e;
            float xi0 = 0.f, xi1 = 0.f, h0 = 0.f, h1 = 0.f;
            int d = -1;
            if (eg < nEdges) {
                const int s = srcIdx[eg];
                d           = dstIdx[eg];
                const float2 xj = *(const float2*)(x + (size_t)s * DIM + 2 * lane);
                const float2 xi = *(const float2*)(x + (size_t)d * DIM + 2 * lane);
                xi0 = xi.x; xi1 = xi.y;
                h0 = xj.x - xi.x; h1 = xj.y - xi.y;
            }
            v2h pi = { (_Float16)xi0, (_Float16)xi1 };
            v2h ph = { (_Float16)h0,  (_Float16)h1  };
            *(v2h*)(A + e * K1 + 2 * lane)       = pi;   // ds_store_b32
            *(v2h*)(A + e * K1 + DIM + 2 * lane) = ph;   // ds_store_b32
            if (lane == 0) dstW[e] = d;
        }
        asm volatile("s_wait_dscnt 0" ::: "memory");     // wave-private LDS RAW

        // ---- GEMM1: H16x64 = relu(A16x128 @ W1^T + b1) ----
        v8f acc[4];
#pragma unroll
        for (int t = 0; t < 4; ++t) {
            const float bb = sB1[t * 16 + m16];          // bias depends on N only
#pragma unroll
            for (int r = 0; r < 8; ++r) acc[t][r] = bb;
        }
#pragma unroll
        for (int ks = 0; ks < 4; ++ks) {
            const v16h a = load_a_frag(A + m16 * K1 + ks * 32 + kq * 8);
#pragma unroll
            for (int t = 0; t < 4; ++t) {
                const v16h b = load_b_frag(sW1 + (t * 16 + m16) * K1 + ks * 32 + kq * 8);
                acc[t] = wmma16(a, b, acc[t]);
            }
        }

        // ReLU -> f16 hidden in LDS (row-major 16x64), next A operand.
#pragma unroll
        for (int t = 0; t < 4; ++t) {
#pragma unroll
            for (int r = 0; r < 8; ++r) {
                const int m = r + 8 * kq;                // C/D: M = r + 8*(lane>=16)
                const int n = t * 16 + m16;              //      N = lane&15 (+tile)
                float v = acc[t][r];
                v = v > 0.f ? v : 0.f;
                H[m * DIM + n] = (_Float16)v;
            }
        }
        asm volatile("s_wait_dscnt 0" ::: "memory");

        // ---- GEMM2: M16x64 = H16x64 @ W2^T + b2 ----
#pragma unroll
        for (int t = 0; t < 4; ++t) {
            const float bb = sB2[t * 16 + m16];
#pragma unroll
            for (int r = 0; r < 8; ++r) acc[t][r] = bb;
        }
#pragma unroll
        for (int ks = 0; ks < 2; ++ks) {
            const v16h a = load_a_frag(H + m16 * DIM + ks * 32 + kq * 8);
#pragma unroll
            for (int t = 0; t < 4; ++t) {
                const v16h b = load_b_frag(sW2 + (t * 16 + m16) * DIM + ks * 32 + kq * 8);
                acc[t] = wmma16(a, b, acc[t]);
            }
        }

        // ---- segment-max scatter (monotone int encoding -> atomicMax i32) ----
#pragma unroll
        for (int r = 0; r < 8; ++r) {
            const int m = r + 8 * kq;
            const int d = dstW[m];
            if (d >= 0) {
#pragma unroll
                for (int t = 0; t < 4; ++t) {
                    const int n = t * 16 + m16;
                    atomicMax(aggEnc + (size_t)d * DIM + n, enc_f32(acc[t][r]));
                }
            }
        }
    }
}

__global__ void init_enc_kernel(int* __restrict__ p, int n) {
    int i = blockIdx.x * blockDim.x + threadIdx.x;
    if (i < n) p[i] = ENC_NEG_INF;
}

// Decode; untouched segments (still enc(-inf)) become 0 like PyG max-aggr.
__global__ void finalize_kernel(const int* __restrict__ e, float* __restrict__ out, int n) {
    int i = blockIdx.x * blockDim.x + threadIdx.x;
    if (i < n) {
        const int v = e[i];
        out[i] = (v == ENC_NEG_INF) ? 0.f : dec_f32(v);
    }
}

extern "C" void kernel_launch(void* const* d_in, const int* in_sizes, int n_in,
                              void* d_out, int out_size, void* d_ws, size_t ws_size,
                              hipStream_t stream) {
    const float* x   = (const float*)d_in[0];
    const int*   ei  = (const int*)d_in[1];      // [2, E] row-major: src then dst
    const float* W1a = (const float*)d_in[2];
    const float* b1a = (const float*)d_in[3];
    const float* W2a = (const float*)d_in[4];
    const float* b2a = (const float*)d_in[5];
    const float* W1b = (const float*)d_in[6];
    const float* b1b = (const float*)d_in[7];
    const float* W2b = (const float*)d_in[8];
    const float* b2b = (const float*)d_in[9];

    const int E = in_sizes[1] / 2;
    const int N = in_sizes[0] / DIM;
    const int nElem = N * DIM;

    const int* src = ei;
    const int* dst = ei + E;

    int*   agg1 = (int*)d_ws;                                   // N*DIM ints
    float* h    = (float*)((char*)d_ws + (size_t)nElem * 4);    // N*DIM floats

    const int tiles      = (E + 15) / 16;
    const int tilesPerBk = WAVES * TILES_PER_WAVE;
    const int cBlocks    = (tiles + tilesPerBk - 1) / tilesPerBk;
    const int eBlocks    = (nElem + 255) / 256;

    // Layer 1: x -> agg1 -> h
    init_enc_kernel<<<eBlocks, 256, 0, stream>>>(agg1, nElem);
    devconv_kernel<<<cBlocks, TPB, 0, stream>>>(x, src, dst, W1a, b1a, W2a, b2a, agg1, E);
    finalize_kernel<<<eBlocks, 256, 0, stream>>>(agg1, h, nElem);

    // Layer 2: h -> d_out (encoded in place, then decoded in place)
    int* agg2 = (int*)d_out;
    init_enc_kernel<<<eBlocks, 256, 0, stream>>>(agg2, nElem);
    devconv_kernel<<<cBlocks, TPB, 0, stream>>>(h, src, dst, W1b, b1b, W2b, b2b, agg2, E);
    finalize_kernel<<<eBlocks, 256, 0, stream>>>((const int*)d_out, (float*)d_out, nElem);
}
